// HybridThoughtAwareAttentionTPU_37915971289441
// MI455X (gfx1250) — matrix operand order repaired and checked by reading, not compiled, not run on hardware
//
#include <hip/hip_runtime.h>
#include <stdint.h>

typedef unsigned short u16;
typedef __attribute__((ext_vector_type(16))) __bf16 v16bf;
typedef __attribute__((ext_vector_type(8)))  float  v8f;
typedef __attribute__((ext_vector_type(4)))  uint32_t u32x4;
typedef __attribute__((ext_vector_type(8)))  uint32_t u32x8;

union FragAB { v16bf v; uint4 q[2]; };

constexpr int B_ = 2, S_ = 2048, D_ = 1024, H_ = 16, T_ = 8, DH = 64;
constexpr float SCALE    = 0.125f;            // 1/sqrt(64)
constexpr float INV_KEEP = 1.0f / 0.9f;       // dropout rate 0.1
constexpr uint32_t KEEP_THRESH = 3865470566u; // floor(0.9 * 2^32)

#define DEV __device__ __forceinline__

DEV uint32_t hash32(uint32_t x) {
  x ^= x >> 16; x *= 0x7feb352du;
  x ^= x >> 15; x *= 0x846ca68bu;
  x ^= x >> 16; return x;
}
DEV bool keep_drop(uint32_t idx, uint32_t seed) {
  return hash32(idx ^ seed) < KEEP_THRESH;
}

DEV float b2f(u16 x) {
  union { uint32_t u; float f; } c; c.u = ((uint32_t)x) << 16; return c.f;
}
DEV u16 f2b(float f) {
  union { float f; uint32_t u; } c; c.f = f;
  uint32_t u = c.u + 0x7FFFu + ((c.u >> 16) & 1u);  // round-to-nearest-even
  return (u16)(u >> 16);
}

DEV v8f wmma_bf16(const v16bf& a, const v16bf& b, const v8f& c) {
  return __builtin_amdgcn_wmma_f32_16x16x32_bf16(
      false, a, false, b, (short)0, c, false, false);
}

// CDNA5 async global->LDS copy (16B per lane), tracked by ASYNCcnt.
DEV void async_b128(void* lds, const void* g) {
  asm volatile("global_load_async_to_lds_b128 %0, %1, off"
               :: "v"((uint32_t)(uintptr_t)lds), "v"(g)
               : "memory");
}
DEV void wait_async_all() {
  asm volatile("s_wait_asynccnt 0x0" ::: "memory");
}

// CDNA5 Tensor Data Mover: 2D tile DMA global->LDS, descriptor in SGPRs
// (D# groups 0 and 1; VADDR2/3 omitted => up-to-2D tensor). TENSORcnt-tracked.
DEV void tdm_load_2d(u32x4 g0, u32x8 g1) {
  asm volatile("tensor_load_to_lds %0, %1" :: "s"(g0), "s"(g1) : "memory");
}
DEV void wait_tensor_all() {
  __builtin_amdgcn_s_wait_tensorcnt(0);
}

// Build D# group0: count=1, lds byte address, 57-bit global address, type=2.
DEV u32x4 tdm_group0(const void* gaddr, void* lds) {
  uint64_t ga = (uint64_t)(uintptr_t)gaddr;
  u32x4 g0;
  g0.x = 1u;                                   // count=1, is_restore=0, gather=0
  g0.y = (uint32_t)(uintptr_t)lds;             // lds_addr (bytes)
  g0.z = (uint32_t)ga;                         // global_addr[31:0]
  g0.w = (uint32_t)((ga >> 32) & 0x01FFFFFFu)  // global_addr[56:32]
         | 0x80000000u;                        // type=2 ("image") in bits 127:126
  return g0;
}
// Build D# group1 for a 64x64 bf16 tile with row stride `stride` (elements),
// LDS padding of 4 DWORDs (8 elems) after every 32 DWORDs (64 elems) => 72-elem rows.
DEV u32x8 tdm_group1_64x64(uint32_t stride) {
  u32x8 g1;
  g1[0] = (1u << 16)        // data_size = 1 -> 2 bytes
        | (1u << 20)        // pad_enable
        | (4u << 22)        // pad_interval: 32 DWORDs (128B = one 64-elem row)
        | (3u << 25);       // pad_amount: 4 DWORDs (16B = 8 elems)
  g1[1] = 64u << 16;        // tensor_dim0[15:0] = 64 (atomic_barrier_addr = 0)
  g1[2] = 64u << 16;        // tensor_dim0[31:16]=0, tensor_dim1[15:0] = 64
  g1[3] = 64u << 16;        // tensor_dim1[31:16]=0, tile_dim0 = 64
  g1[4] = 64u;              // tile_dim1 = 64, tile_dim2 = 0
  g1[5] = stride;           // tensor_dim0_stride[31:0]
  g1[6] = 0u;               // stride[47:32]=0, tensor_dim1_stride[15:0]=0
  g1[7] = 0u;
  return g1;
}

// ---------------------------------------------------------------- converters
__global__ void f2bf_kernel(const float* __restrict__ in, u16* __restrict__ out, int n4) {
  int i = blockIdx.x * blockDim.x + threadIdx.x;
  if (i < n4) {
    float4 f = ((const float4*)in)[i];
    ushort4 o;
    o.x = f2b(f.x); o.y = f2b(f.y); o.z = f2b(f.z); o.w = f2b(f.w);
    ((ushort4*)out)[i] = o;
  }
}

// in[K][N] fp32 -> out[N][K] bf16 (pre-transposed weights), tiled via LDS.
__global__ __launch_bounds__(256) void f2bf_t_kernel(
    const float* __restrict__ in, u16* __restrict__ out, int K, int N) {
  __shared__ float tile[32][33];
  int k0 = blockIdx.y * 32, n0 = blockIdx.x * 32;
  int tx = threadIdx.x & 31, ty = threadIdx.x >> 5;  // 32 x 8
#pragma unroll
  for (int i = 0; i < 32; i += 8)
    tile[ty + i][tx] = in[(size_t)(k0 + ty + i) * N + n0 + tx];
  __syncthreads();
#pragma unroll
  for (int i = 0; i < 32; i += 8)
    out[(size_t)(n0 + ty + i) * K + k0 + tx] = f2b(tile[tx][ty + i]);
}

// ---------------------------------------------------------------- WMMA GEMM
// C[M,N] = A[M,K](bf16,rowmajor) * Bt[N,K](bf16, pre-transposed) + bias[N]
// Block tile 128x128, 8 waves in 2x4 grid, each wave 64x32 (4x2 wmma tiles).
// LDS staging via global_load_async_to_lds_b128 (no transposes needed).
// MODE: 0 = fp32 out, 1 = bf16 out, 2 = bf16 out transposed [B][D][S] (for V)
template <int MODE>
__global__ __launch_bounds__(256) void gemm_kernel(
    const u16* __restrict__ A, const u16* __restrict__ Bt_g,
    const float* __restrict__ bias, void* __restrict__ Cout,
    int M, int N, int K) {
  __shared__ alignas(16) u16 As[128][40];  // [m][k], 80B rows (16B multiple)
  __shared__ alignas(16) u16 Bt[128][40];  // [n][k]

  const int tid = threadIdx.x, lane = tid & 31, w = tid >> 5;
  const int wm = (w >> 2) * 64, wn = (w & 3) * 32;
  const int m0 = blockIdx.y * 128, n0 = blockIdx.x * 128;
  const int lm = lane & 15, hi = (lane >> 4) & 1;

  const int sr = tid >> 1, scc = (tid & 1) * 16;  // staging coords

  v8f acc[4][2];
#pragma unroll
  for (int i = 0; i < 4; i++)
#pragma unroll
    for (int j = 0; j < 2; j++)
      acc[i][j] = v8f{0.f, 0.f, 0.f, 0.f, 0.f, 0.f, 0.f, 0.f};

  for (int k0 = 0; k0 < K; k0 += 32) {
    __syncthreads();  // previous iteration's frag reads complete
    if (m0 + sr < M) {
      const u16* ga = A + (size_t)(m0 + sr) * K + k0 + scc;
      async_b128(&As[sr][scc], ga);
      async_b128(&As[sr][scc + 8], ga + 8);
    }
    {
      const u16* gb = Bt_g + (size_t)(n0 + sr) * K + k0 + scc;
      async_b128(&Bt[sr][scc], gb);
      async_b128(&Bt[sr][scc + 8], gb + 8);
    }
    wait_async_all();
    __syncthreads();

    FragAB a[4], bb[2];
#pragma unroll
    for (int i = 0; i < 4; i++) {  // A frag 16x32: lanes<16 K0-7,16-23; >=16 K8-15,24-31
      int row = wm + i * 16 + lm;
      int klo = hi ? 8 : 0;
      a[i].q[0] = *(const uint4*)&As[row][klo];
      a[i].q[1] = *(const uint4*)&As[row][16 + klo];
    }
#pragma unroll
    for (int j = 0; j < 2; j++) {  // B frag 32x16: lane=n, lanes<16 K0-7,16-23; >=16 K8-15,24-31
      int col = wn + j * 16 + lm;
      int kb = hi ? 8 : 0;
      bb[j].q[0] = *(const uint4*)&Bt[col][kb];
      bb[j].q[1] = *(const uint4*)&Bt[col][kb + 16];
    }
#pragma unroll
    for (int i = 0; i < 4; i++)
#pragma unroll
      for (int j = 0; j < 2; j++)
        acc[i][j] = wmma_bf16(a[i].v, bb[j].v, acc[i][j]);
  }

#pragma unroll
  for (int j = 0; j < 2; j++) {
    int n = n0 + wn + j * 16 + lm;
    float bv = bias ? bias[n] : 0.f;
#pragma unroll
    for (int i = 0; i < 4; i++) {
#pragma unroll
      for (int r = 0; r < 8; r++) {
        int m = m0 + wm + i * 16 + r + (hi ? 8 : 0);
        if (m < M) {
          float val = acc[i][j][r] + bv;
          if (MODE == 0) {
            ((float*)Cout)[(size_t)m * N + n] = val;
          } else if (MODE == 1) {
            ((u16*)Cout)[(size_t)m * N + n] = f2b(val);
          } else {  // transposed [B][D][S]: m = b*S + s, n = d
            int b = m >> 11, s = m & (S_ - 1);
            ((u16*)Cout)[((size_t)b * D_ + n) * S_ + s] = f2b(val);
          }
        }
      }
    }
  }
}

// ---------------------------------------------------------------- flash attention
// grid (S/128, H, B), 256 threads; wave w handles 16 query rows.
// K tile [key][dh] and V tile [dh][key] staged by the Tensor Data Mover:
// one 64x64 bf16 2D-tile descriptor each, LDS padding 8 elems / 64 elems
// (72-elem rows), issued by wave 0, completion via s_wait_tensorcnt.
__global__ __launch_bounds__(256) void flash_kernel(
    const u16* __restrict__ q, const u16* __restrict__ k,
    const u16* __restrict__ vt, const unsigned char* __restrict__ mask,
    float* __restrict__ out) {
  __shared__ alignas(16) u16 Ks[64][72];     // [key][dh], 144B rows
  __shared__ alignas(16) u16 Vt[64][72];     // [dh][key]
  __shared__ alignas(16) u16 Ps[8][16][72];  // per-wave P staging [row][key]

  const int tid = threadIdx.x, lane = tid & 31, w = tid >> 5;
  const int b = blockIdx.z, h = blockIdx.y, q0 = blockIdx.x * 128;
  const int qr0 = q0 + w * 16;
  const int lm = lane & 15, hi = (lane >> 4) & 1;

  // Q fragments (A layout), resident for the whole loop
  FragAB qf[2];
  {
    int row = qr0 + lm, klo = hi ? 8 : 0;
    const u16* qp = q + ((size_t)(b * S_ + row)) * D_ + h * DH;
    qf[0].q[0] = *(const uint4*)(qp + klo);
    qf[0].q[1] = *(const uint4*)(qp + 16 + klo);
    qf[1].q[0] = *(const uint4*)(qp + 32 + klo);
    qf[1].q[1] = *(const uint4*)(qp + 48 + klo);
  }

  float mrow[8], lsum[8];
  v8f o[4];
#pragma unroll
  for (int r = 0; r < 8; r++) { mrow[r] = -INFINITY; lsum[r] = 0.f; }
#pragma unroll
  for (int c = 0; c < 4; c++) o[c] = v8f{0.f, 0.f, 0.f, 0.f, 0.f, 0.f, 0.f, 0.f};

  for (int kc = 0; kc < S_; kc += 64) {
    __syncthreads();
    if (w == 0) {  // TDM: stage both tiles with two descriptors
      const u16* gk = k + ((size_t)(b * S_ + kc)) * D_ + h * DH;
      const u16* gv = vt + ((size_t)b * D_ + h * DH) * S_ + kc;
      tdm_load_2d(tdm_group0(gk, &Ks[0][0]), tdm_group1_64x64((uint32_t)D_));
      tdm_load_2d(tdm_group0(gv, &Vt[0][0]), tdm_group1_64x64((uint32_t)S_));
      wait_tensor_all();
    }
    __syncthreads();

    // S = Q * K^T for 4 groups of 16 keys (2 wmma each over dh=64)
    v8f sc[4];
#pragma unroll
    for (int g = 0; g < 4; g++) {
      sc[g] = v8f{0.f, 0.f, 0.f, 0.f, 0.f, 0.f, 0.f, 0.f};
#pragma unroll
      for (int dhalf = 0; dhalf < 2; dhalf++) {
        FragAB kf;
        int row = g * 16 + lm, col = dhalf * 32 + (hi ? 8 : 0);
        kf.q[0] = *(const uint4*)&Ks[row][col];
        kf.q[1] = *(const uint4*)&Ks[row][col + 16];
        sc[g] = wmma_bf16(qf[dhalf].v, kf.v, sc[g]);
      }
    }

    // scale + key mask
#pragma unroll
    for (int g = 0; g < 4; g++) {
      int key = kc + g * 16 + lm;
      bool mk = mask[b * S_ + key] != 0;
#pragma unroll
      for (int r = 0; r < 8; r++) sc[g][r] = mk ? sc[g][r] * SCALE : -1e9f;
    }

    // chunk row-max (reduce across 16 lanes of each half)
    float cmax[8];
#pragma unroll
    for (int r = 0; r < 8; r++) {
      float m4 = fmaxf(fmaxf(sc[0][r], sc[1][r]), fmaxf(sc[2][r], sc[3][r]));
      m4 = fmaxf(m4, __shfl_xor(m4, 1, 32));
      m4 = fmaxf(m4, __shfl_xor(m4, 2, 32));
      m4 = fmaxf(m4, __shfl_xor(m4, 4, 32));
      m4 = fmaxf(m4, __shfl_xor(m4, 8, 32));
      cmax[r] = m4;
    }

    // online softmax rescale
#pragma unroll
    for (int r = 0; r < 8; r++) {
      float nm = fmaxf(mrow[r], cmax[r]);
      float corr = __expf(mrow[r] - nm);
      mrow[r] = nm;
      lsum[r] *= corr;
#pragma unroll
      for (int c = 0; c < 4; c++) o[c][r] *= corr;
    }

    // p = exp(s - m), accumulate denom (pre-dropout), stage dropped P as bf16
    const int qbase = qr0 + (hi ? 8 : 0);
#pragma unroll
    for (int g = 0; g < 4; g++) {
      int key = kc + g * 16 + lm;
#pragma unroll
      for (int r = 0; r < 8; r++) {
        float p = __expf(sc[g][r] - mrow[r]);
        lsum[r] += p;
        uint32_t idx = (uint32_t)((((b * H_ + h) * S_ + (qbase + r)) << 11) + key);
        float pv = keep_drop(idx, 0x9e3779b9u) ? p * INV_KEEP : 0.f;
        Ps[w][r + (hi ? 8 : 0)][g * 16 + lm] = f2b(pv);
      }
    }

    // O += P * V   (contraction over 64 keys, two A frags x four dh tiles)
#pragma unroll
    for (int kg = 0; kg < 2; kg++) {
      FragAB pa;
      int kbase = kg * 32 + (hi ? 8 : 0);
      pa.q[0] = *(const uint4*)&Ps[w][lm][kbase];
      pa.q[1] = *(const uint4*)&Ps[w][lm][kbase + 16];
#pragma unroll
      for (int c = 0; c < 4; c++) {
        FragAB vf;
        int vrow = c * 16 + lm, kb = kg * 32 + (hi ? 8 : 0);
        vf.q[0] = *(const uint4*)&Vt[vrow][kb];
        vf.q[1] = *(const uint4*)&Vt[vrow][kb + 16];
        o[c] = wmma_bf16(pa.v, vf.v, o[c]);
      }
    }
  }

  // finalize: reduce denom across half-lanes, normalize, store fp32
  float rinv[8];
#pragma unroll
  for (int r = 0; r < 8; r++) {
    float s = lsum[r];
    s += __shfl_xor(s, 1, 32);
    s += __shfl_xor(s, 2, 32);
    s += __shfl_xor(s, 4, 32);
    s += __shfl_xor(s, 8, 32);
    rinv[r] = s > 0.f ? 1.f / s : 0.f;
  }
#pragma unroll
  for (int c = 0; c < 4; c++) {
#pragma unroll
    for (int r = 0; r < 8; r++) {
      int row = qr0 + r + (hi ? 8 : 0);
      out[((size_t)(b * S_ + row)) * D_ + h * DH + c * 16 + lm] = o[c][r] * rinv[r];
    }
  }
}

// ---------------------------------------------------------------- thought attention (Skv=8)
__global__ void thought_kernel(const u16* __restrict__ tq, const u16* __restrict__ tk,
                               const u16* __restrict__ tv, float* __restrict__ out) {
  int gid = blockIdx.x * blockDim.x + threadIdx.x;
  if (gid >= B_ * S_ * H_) return;
  int h = gid & (H_ - 1), row = gid >> 4;
  const u16* qp = tq + (size_t)row * D_ + h * DH;

  float s[T_];
  float smax = -1e30f;
#pragma unroll
  for (int j = 0; j < T_; j++) {
    const u16* kp = tk + (size_t)j * D_ + h * DH;
    float a = 0.f;
    for (int d = 0; d < DH; d++) a += b2f(qp[d]) * b2f(kp[d]);
    a *= SCALE;
    s[j] = a;
    smax = fmaxf(smax, a);
  }
  float ssum = 0.f;
#pragma unroll
  for (int j = 0; j < T_; j++) { s[j] = __expf(s[j] - smax); ssum += s[j]; }
  float inv = 1.f / ssum;
#pragma unroll
  for (int j = 0; j < T_; j++) {
    float wv = s[j] * inv;
    uint32_t idx = ((uint32_t)gid << 3) + j;
    s[j] = keep_drop(idx, 0x85ebca6bu) ? wv * INV_KEEP : 0.f;
  }
  float* op = out + (size_t)row * D_ + h * DH;
  for (int d = 0; d < DH; d++) {
    float a = 0.f;
#pragma unroll
    for (int j = 0; j < T_; j++) a += s[j] * b2f(tv[(size_t)j * D_ + h * DH + d]);
    op[d] += a;
  }
}

// ---------------------------------------------------------------- launch
extern "C" void kernel_launch(void* const* d_in, const int* in_sizes, int n_in,
                              void* d_out, int out_size, void* d_ws, size_t ws_size,
                              hipStream_t stream) {
  const float* x  = (const float*)d_in[0];
  const unsigned char* mask = (const unsigned char*)d_in[1];
  const float* ts = (const float*)d_in[2];
  const float* Wq = (const float*)d_in[3];  const float* bq  = (const float*)d_in[4];
  const float* Wk = (const float*)d_in[5];  const float* bk  = (const float*)d_in[6];
  const float* Wv = (const float*)d_in[7];  const float* bv  = (const float*)d_in[8];
  const float* Wtq = (const float*)d_in[9]; const float* btq = (const float*)d_in[10];
  const float* Wtk = (const float*)d_in[11];const float* btk = (const float*)d_in[12];
  const float* Wtv = (const float*)d_in[13];const float* btv = (const float*)d_in[14];
  const float* Wo = (const float*)d_in[15]; const float* bo  = (const float*)d_in[16];
  float* out = (float*)d_out;

  const int MS = B_ * S_;                   // 4096
  const size_t NX = (size_t)MS * D_;        // 4,194,304
  const size_t NW = (size_t)D_ * D_;        // 1,048,576
  const size_t NT = (size_t)T_ * D_;        // 8,192

  char* ws = (char*)d_ws;
  size_t off = 0;
  auto alloc = [&](size_t bytes) {
    void* p = ws + off;
    off += (bytes + 255) & ~(size_t)255;
    return p;
  };
  u16* xb   = (u16*)alloc(NX * 2);
  u16* tsb  = (u16*)alloc(NT * 2);
  u16* Wqt  = (u16*)alloc(NW * 2);   // all weights stored transposed [N][K] bf16
  u16* Wkt  = (u16*)alloc(NW * 2);
  u16* Wvt  = (u16*)alloc(NW * 2);
  u16* Wtqt = (u16*)alloc(NW * 2);
  u16* Wtkt = (u16*)alloc(NW * 2);
  u16* Wtvt = (u16*)alloc(NW * 2);
  u16* Wot  = (u16*)alloc(NW * 2);
  u16* qb   = (u16*)alloc(NX * 2);
  u16* kb   = (u16*)alloc(NX * 2);
  u16* vtb  = (u16*)alloc(NX * 2);   // V pre-transposed: [B][D][S] bf16
  u16* tqb  = (u16*)alloc(NX * 2);
  u16* tkb  = (u16*)alloc(NT * 2);
  u16* tvb  = (u16*)alloc(NT * 2);
  float* attnf = (float*)alloc(NX * 4);
  u16* attnb   = (u16*)alloc(NX * 2);

  auto cvt = [&](const float* src, u16* dst, size_t n) {
    int n4 = (int)(n / 4);
    f2bf_kernel<<<(n4 + 255) / 256, 256, 0, stream>>>(src, dst, n4);
  };
  auto cvtT = [&](const float* src, u16* dst) {  // [D,D] -> transposed bf16
    dim3 g(D_ / 32, D_ / 32);
    f2bf_t_kernel<<<g, 256, 0, stream>>>(src, dst, D_, D_);
  };
  cvt(x, xb, NX);
  cvt(ts, tsb, NT);
  cvtT(Wq, Wqt);  cvtT(Wk, Wkt);  cvtT(Wv, Wvt);
  cvtT(Wtq, Wtqt); cvtT(Wtk, Wtkt); cvtT(Wtv, Wtvt);
  cvtT(Wo, Wot);

  dim3 gproj(D_ / 128, MS / 128);   // (8, 32)
  gemm_kernel<1><<<gproj, 256, 0, stream>>>(xb, Wqt,  bq,  qb,  MS, D_, D_);
  gemm_kernel<1><<<gproj, 256, 0, stream>>>(xb, Wkt,  bk,  kb,  MS, D_, D_);
  gemm_kernel<2><<<gproj, 256, 0, stream>>>(xb, Wvt,  bv,  vtb, MS, D_, D_);
  gemm_kernel<1><<<gproj, 256, 0, stream>>>(xb, Wtqt, btq, tqb, MS, D_, D_);

  dim3 gth(D_ / 128, 1);
  gemm_kernel<1><<<gth, 256, 0, stream>>>(tsb, Wtkt, btk, tkb, T_, D_, D_);
  gemm_kernel<1><<<gth, 256, 0, stream>>>(tsb, Wtvt, btv, tvb, T_, D_, D_);

  dim3 gfa(S_ / 128, H_, B_);       // (16, 16, 2)
  flash_kernel<<<gfa, 256, 0, stream>>>(qb, kb, vtb, mask, attnf);

  thought_kernel<<<(B_ * S_ * H_) / 256, 256, 0, stream>>>(tqb, tkb, tvb, attnf);

  cvt(attnf, attnb, NX);
  gemm_kernel<0><<<gproj, 256, 0, stream>>>(attnb, Wot, bo, out, MS, D_, D_);
}
